// MM_GFN_23562190586341
// MI455X (gfx1250) — matrix operand order repaired and checked by reading, compiled, not run on hardware
//
#include <hip/hip_runtime.h>
#include <hip/hip_bf16.h>
#include <math.h>

// ---------------------------------------------------------------------------
// MI455X (gfx1250) implementation. wave32, WMMA bf16 16x16x32 (f32 accum).
// Compute-bound workload (~90 GFLOPs, ~300MB HBM) -> all GEMMs via WMMA.
// GRU recurrence is batch-parallel: combined [Wih^T;Whh^T] resident in LDS
// (144KB bf16), hidden state in registers for all 48 steps, per-step x tile
// pre-packed into WMMA-A layout in global and DMA'd into LDS with the
// Tensor Data Mover (TENSORcnt-synchronized).
// ---------------------------------------------------------------------------

#define BTOT 8192
#define TSTEPS 48

typedef __bf16 bf16t;
typedef __attribute__((ext_vector_type(16))) __bf16 v16bf;
typedef __attribute__((ext_vector_type(8)))  float v8f;
typedef __attribute__((ext_vector_type(4)))  unsigned int u32x4;
typedef __attribute__((ext_vector_type(8)))  int i32x8;
typedef __attribute__((ext_vector_type(4)))  int i32x4;

#if defined(__has_builtin)
#if __has_builtin(__builtin_amdgcn_tensor_load_to_lds) && __has_builtin(__builtin_amdgcn_s_wait_tensorcnt)
#define HAVE_TDM 1
#endif
#endif
#ifndef HAVE_TDM
#define HAVE_TDM 0
#endif

__device__ __forceinline__ float act_apply(float x, int act) {
  if (act == 1) return 0.5f * x * (1.0f + erff(x * 0.70710678118654752f)); // exact gelu
  if (act == 2) return 1.0f / (1.0f + expf(-x));                          // sigmoid
  if (act == 3) return fmaxf(x, 0.0f);                                    // relu
  return x;
}

// WMMA operand lane layouts (ISA 7.12.2, bf16 16x16x32):
//  A: lane L: m=L%16, half=L/16; elem e<8 -> k=e+8*half ; e>=8 -> k=e+8+8*half
//  B: lane L: n=L%16, half=L/16; elem e<8 -> k=e+8*half ; e>=8 -> k=e+8+8*half
//  C: lane L: n=L%16; vgpr v   -> m=v+8*(L/16)

#if HAVE_TDM
// Issue a TDM copy of one contiguous 4KB block (512 x 8B) from global to LDS.
// D# layout per CDNA5 ISA (group0: count/lds/global/type,
// group1: data_size=8B, tensor_dim0=tile_dim0=512, stride=512).
// This toolchain's builtin takes 6 args (g0,g1,g2,g3,g4,cpol).
__device__ __forceinline__ void tdm_copy4k(unsigned int lds_off, const bf16t* gsrc) {
  unsigned long long ga = (unsigned long long)(uintptr_t)gsrc;
  u32x4 g0;
  g0.x = 1u;                                                    // count=1, user D#
  g0.y = lds_off;                                               // lds_addr (bytes)
  g0.z = (unsigned int)ga;                                      // global_addr[31:0]
  g0.w = (unsigned int)((ga >> 32) & 0x01FFFFFFull) | (2u << 30); // addr[56:32] | type=2
  i32x8 g1;
  g1[0] = (3 << 16);            // data_size = 8B; no multicast/iterate/pad
  g1[1] = (int)(512u << 16);    // tensor_dim0[15:0] = 512 (in 8B units)
  g1[2] = 0;                    // tensor_dim0 hi / tensor_dim1 lo
  g1[3] = (int)(512u << 16);    // tile_dim0 = 512
  g1[4] = 0;                    // tile_dim1 = tile_dim2 = 0 (1-D)
  g1[5] = 512;                  // tensor_dim0_stride = 512
  g1[6] = 0;
  g1[7] = 0;
  i32x4 z4 = (i32x4){0, 0, 0, 0};
  i32x8 z8 = (i32x8){0, 0, 0, 0, 0, 0, 0, 0};
  __builtin_amdgcn_tensor_load_to_lds(g0, g1, z4, z4, z8, 0);
}
#endif

// ---------------------------------------------------------------------------
// Pack ts_x[:, :, :64] into WMMA-A operand blocks:
//   xp[((rb*48 + t)*4 + ab)*512 + L*16 + e]  (bf16), ab = kb*2 + rt
// One (rb,t) slot is a contiguous 4KB block -> per-step GRU staging is 1 DMA.
// ---------------------------------------------------------------------------
__global__ __launch_bounds__(256) void pack_x_kernel(const float* __restrict__ tsX,
                                                     bf16t* __restrict__ xp)
{
  int i = blockIdx.x * 256 + threadIdx.x;      // [0, 256*48*2048)
  int rem = i & 511, L = rem >> 4, e = rem & 15;
  int ab = (i >> 9) & 3;
  int rbt = i >> 11;
  int t = rbt % TSTEPS;
  int rb = rbt / TSTEPS;
  int kb = ab >> 1, rt = ab & 1;
  int m = rt * 16 + (L & 15), half = L >> 4;
  int k = (kb << 5) + ((e < 8) ? (e + 8 * half) : (e + 8 + 8 * half));
  xp[i] = (bf16t)tsX[(size_t)(rb * 32 + m) * 6144 + t * 128 + k];
}

// ---------------------------------------------------------------------------
// Generic WMMA GEMM: C[M,256] = act(A[M,K] @ W + bias).  M-tile = 32 rows.
// amode: 0 = plain A (lda);  1 = gather xg rows (p0=sd_node,p1=ts_h,p2=text);
//        2 = concat(sd_h|ts_h) along K (p0,p1).
// ---------------------------------------------------------------------------
__global__ __launch_bounds__(256) void gemm256_wmma(
    const float* __restrict__ A, int lda,
    const float* __restrict__ W, int wTrans,
    const float* __restrict__ bias, float* __restrict__ C,
    int K, int act, int amode,
    const float* __restrict__ p0, const float* __restrict__ p1,
    const float* __restrict__ p2)
{
  __shared__ __attribute__((aligned(32))) bf16t sWt[16 * 512];
  __shared__ __attribute__((aligned(32))) bf16t sAt[2 * 512];

  const int tid  = threadIdx.x;
  const int lane = tid & 31;
  const int wave = tid >> 5;
  const int rowBase = blockIdx.x * 32;

  v8f acc[2][2];
#pragma unroll
  for (int r = 0; r < 2; ++r)
#pragma unroll
    for (int c = 0; c < 2; ++c) acc[r][c] = (v8f){0, 0, 0, 0, 0, 0, 0, 0};

  const int nkb = K >> 5;
  for (int kb = 0; kb < nkb; ++kb) {
    __syncthreads();
    for (int idx = tid; idx < 16 * 512; idx += 256) {
      int nt = idx >> 9, rem = idx & 511, L = rem >> 4, e = rem & 15;
      int n = nt * 16 + (L & 15);
      int k = (kb << 5) + ((L >> 4) << 3) + ((e < 8) ? e : e + 8);
      float v = wTrans ? W[n * K + k] : W[k * 256 + n];
      sWt[idx] = (bf16t)v;
    }
    for (int idx = tid; idx < 2 * 512; idx += 256) {
      int rt = idx >> 9, rem = idx & 511, L = rem >> 4, e = rem & 15;
      int m = rt * 16 + (L & 15), half = L >> 4;
      int kl = (e < 8) ? (e + 8 * half) : (e + 8 + 8 * half);
      int row = rowBase + m;
      int k = (kb << 5) + kl;
      float v;
      if (amode == 0) {
        v = A[(size_t)row * lda + k];
      } else if (amode == 1) {
        int b = row / 9, nd = row - b * 9;
        v = (nd == 0) ? p0[(size_t)b * 256 + k]
          : (nd == 1) ? p1[(size_t)b * 256 + k]
                      : p2[((size_t)b * 7 + (nd - 2)) * 256 + k];
      } else {
        v = (k < 256) ? p0[(size_t)row * 256 + k] : p1[(size_t)row * 256 + (k - 256)];
      }
      sAt[idx] = (bf16t)v;
    }
    __syncthreads();
    v16bf a0 = *(const v16bf*)&sAt[lane * 16];
    v16bf a1 = *(const v16bf*)&sAt[512 + lane * 16];
#pragma unroll
    for (int c = 0; c < 2; ++c) {
      int nt = wave * 2 + c;
      v16bf b = *(const v16bf*)&sWt[nt * 512 + lane * 16];
      acc[0][c] = __builtin_amdgcn_wmma_f32_16x16x32_bf16(false, a0, false, b, (short)0, acc[0][c], false, false);
      acc[1][c] = __builtin_amdgcn_wmma_f32_16x16x32_bf16(false, a1, false, b, (short)0, acc[1][c], false, false);
    }
  }
#pragma unroll
  for (int rt = 0; rt < 2; ++rt)
#pragma unroll
    for (int c = 0; c < 2; ++c) {
      int nt = wave * 2 + c;
      int n = nt * 16 + (lane & 15);
      float bv = bias ? bias[n] : 0.0f;
#pragma unroll
      for (int v = 0; v < 8; ++v) {
        int m = rt * 16 + v + 8 * (lane >> 4);
        C[(size_t)(rowBase + m) * 256 + n] = act_apply(acc[rt][c][v] + bv, act);
      }
    }
}

// ---------------------------------------------------------------------------
// Fused GRU. Workgroup owns 32 batch rows, runs all steps. Combined weights
// [Wih^T;Whh^T] (192x384) resident in LDS (bf16 B-layout, 144KB). Wave w owns
// gate columns [16w,16w+16): tiles {w(r), 8+w(z), 16+w(n)} -> cell in regs.
// Per-step x operand arrives pre-packed via TDM into sAx (TENSORcnt).
// ---------------------------------------------------------------------------
__global__ __launch_bounds__(256) void gru_wmma(
    const bf16t* __restrict__ xpack, // packed [256][48][4*512] bf16 A-blocks
    const float* __restrict__ Wih,   // [384, 64]
    const float* __restrict__ Whh,   // [384, 128]
    const float* __restrict__ bih, const float* __restrict__ bhh,
    float* __restrict__ tsH,         // [B, 256]
    int colOff, int t0, int dir, int nsteps)
{
  __shared__ __attribute__((aligned(32))) bf16t sW[6 * 24 * 512]; // 144 KB weights
  __shared__ __attribute__((aligned(32))) bf16t sAh[8 * 512];     // 8 KB h operand
  __shared__ __attribute__((aligned(32))) bf16t sAx[4 * 512];     // 4 KB x operand (DMA dst)

  const int tid  = threadIdx.x;
  const int lane = tid & 31;
  const int wave = tid >> 5;
  const int rb   = blockIdx.x;
  const int rowBase = rb * 32;
  const int jr = wave * 16 + (lane & 15);   // this lane's gate column (0..127)

  const float bsr = bih[jr] + bhh[jr];
  const float bsz = bih[128 + jr] + bhh[128 + jr];
  const float bin = bih[256 + jr];
  const float bhn = bhh[256 + jr];

  float hprev[2][8];
#pragma unroll
  for (int r = 0; r < 2; ++r)
#pragma unroll
    for (int v = 0; v < 8; ++v) hprev[r][v] = 0.0f;

  // preload combined weights into WMMA B layout
  for (int idx = tid; idx < 6 * 24 * 512; idx += 256) {
    int wb = idx >> 9, rem = idx & 511, L = rem >> 4, e = rem & 15;
    int kb = wb / 24, nt = wb - kb * 24;
    int n = nt * 16 + (L & 15);
    int k = (kb << 5) + ((L >> 4) << 3) + ((e < 8) ? e : e + 8);
    float v = (k < 64) ? Wih[n * 64 + k] : Whh[n * 128 + (k - 64)];
    sW[idx] = (bf16t)v;
  }
  for (int i = tid; i < 8 * 512; i += 256) sAh[i] = (bf16t)0.0f;

  // stage x(t0): TDM DMA of one contiguous 4KB packed block
#if HAVE_TDM
  if (wave == 0 && lane == 0)
    tdm_copy4k((unsigned int)(uintptr_t)&sAx[0],
               xpack + ((size_t)(rb * TSTEPS + t0) << 11));
#else
  for (int i = tid; i < 4 * 512; i += 256)
    sAx[i] = xpack[((size_t)(rb * TSTEPS + t0) << 11) + i];
#endif
  __syncthreads();

  for (int s = 0; s < nsteps; ++s) {
#if HAVE_TDM
    if (wave == 0) __builtin_amdgcn_s_wait_tensorcnt(0);  // x DMA complete
#endif
    __syncthreads();

    v8f accR[2], accZ[2], accNi[2], accNh[2];
#pragma unroll
    for (int rt = 0; rt < 2; ++rt) {
      v16bf a[6];
#pragma unroll
      for (int kb = 0; kb < 2; ++kb)
        a[kb] = *(const v16bf*)&sAx[(kb * 2 + rt) * 512 + lane * 16];
#pragma unroll
      for (int kb = 2; kb < 6; ++kb)
        a[kb] = *(const v16bf*)&sAh[((kb - 2) * 2 + rt) * 512 + lane * 16];

      v8f cr = (v8f){0,0,0,0,0,0,0,0}, cz = cr, cni = cr, cnh = cr;
#pragma unroll
      for (int kb = 0; kb < 6; ++kb) {  // r gate: gi+gh fused
        v16bf b = *(const v16bf*)&sW[(kb * 24 + wave) * 512 + lane * 16];
        cr = __builtin_amdgcn_wmma_f32_16x16x32_bf16(false, a[kb], false, b, (short)0, cr, false, false);
      }
#pragma unroll
      for (int kb = 0; kb < 6; ++kb) {  // z gate
        v16bf b = *(const v16bf*)&sW[(kb * 24 + 8 + wave) * 512 + lane * 16];
        cz = __builtin_amdgcn_wmma_f32_16x16x32_bf16(false, a[kb], false, b, (short)0, cz, false, false);
      }
#pragma unroll
      for (int kb = 0; kb < 2; ++kb) {  // n gate, input part
        v16bf b = *(const v16bf*)&sW[(kb * 24 + 16 + wave) * 512 + lane * 16];
        cni = __builtin_amdgcn_wmma_f32_16x16x32_bf16(false, a[kb], false, b, (short)0, cni, false, false);
      }
#pragma unroll
      for (int kb = 2; kb < 6; ++kb) {  // n gate, hidden part (r * h_n)
        v16bf b = *(const v16bf*)&sW[(kb * 24 + 16 + wave) * 512 + lane * 16];
        cnh = __builtin_amdgcn_wmma_f32_16x16x32_bf16(false, a[kb], false, b, (short)0, cnh, false, false);
      }
      accR[rt] = cr; accZ[rt] = cz; accNi[rt] = cni; accNh[rt] = cnh;
    }
    __syncthreads();   // all waves done reading sAx/sAh

    if (s + 1 < nsteps) {
      int tn = t0 + (s + 1) * dir;
#if HAVE_TDM
      if (wave == 0 && lane == 0)   // overlap next-x DMA with cell phase
        tdm_copy4k((unsigned int)(uintptr_t)&sAx[0],
                   xpack + ((size_t)(rb * TSTEPS + tn) << 11));
      if (wave == 0 && s + 2 < nsteps) {  // prefetch t+2 block into cache
        const char* pf = (const char*)(xpack + ((size_t)(rb * TSTEPS + tn + dir) << 11));
        __builtin_prefetch(pf + lane * 128, 0, 0);
      }
#else
      for (int i = tid; i < 4 * 512; i += 256)
        sAx[i] = xpack[((size_t)(rb * TSTEPS + tn) << 11) + i];
#endif
    }

    // GRU cell in registers + write new h into A-operand layout
#pragma unroll
    for (int rt = 0; rt < 2; ++rt) {
#pragma unroll
      for (int v = 0; v < 8; ++v) {
        float r  = 1.0f / (1.0f + expf(-(accR[rt][v] + bsr)));
        float z  = 1.0f / (1.0f + expf(-(accZ[rt][v] + bsz)));
        float nn = tanhf(accNi[rt][v] + bin + r * (accNh[rt][v] + bhn));
        float h2 = (1.0f - z) * nn + z * hprev[rt][v];
        hprev[rt][v] = h2;
        int m  = rt * 16 + v + 8 * (lane >> 4);
        int kl = jr & 31, half, e;
        if      (kl <  8) { half = 0; e = kl;      }
        else if (kl < 16) { half = 1; e = kl - 8;  }
        else if (kl < 24) { half = 0; e = kl - 8;  }
        else              { half = 1; e = kl - 16; }
        int ab = (jr >> 5) * 2 + (m >> 4);
        int L  = half * 16 + (m & 15);
        sAh[ab * 512 + L * 16 + e] = (bf16t)h2;
      }
    }
    __syncthreads();
  }
  // write final hidden state
#pragma unroll
  for (int rt = 0; rt < 2; ++rt)
#pragma unroll
    for (int v = 0; v < 8; ++v) {
      int m = rt * 16 + v + 8 * (lane >> 4);
      tsH[(size_t)(rowBase + m) * 256 + colOff + jr] = hprev[rt][v];
    }
}

// txt_rep = (ts_h + sum(text_emb)) / 8   (mean of input nodes 1..8)
__global__ void txt_rep_kernel(const float* __restrict__ tsH,
                               const float* __restrict__ text,
                               float* __restrict__ txt)
{
  int i = blockIdx.x * blockDim.x + threadIdx.x;
  if (i >= BTOT * 256) return;
  int b = i >> 8, j = i & 255;
  float acc = tsH[i];
#pragma unroll
  for (int n = 0; n < 7; ++n) acc += text[((size_t)b * 7 + n) * 256 + j];
  txt[i] = acc * 0.125f;
}

// GAT layer-1 attention (fully-connected 9 nodes + self loops): thread per (sample, head)
__global__ void gat1_attn(const float* __restrict__ XP, const float* __restrict__ aSrc,
                          const float* __restrict__ aDst, const float* __restrict__ b1,
                          float* __restrict__ M1)
{
  int g = blockIdx.x * blockDim.x + threadIdx.x;
  if (g >= BTOT * 8) return;
  int b = g >> 3, h = g & 7;
  const float* base = XP + (size_t)b * 9 * 256 + h * 32;
  float es[9], ed[9];
#pragma unroll
  for (int s = 0; s < 9; ++s) {
    float a = 0.f, d = 0.f;
    for (int k = 0; k < 32; ++k) {
      float v = base[s * 256 + k];
      a += v * aSrc[h * 32 + k];
      d += v * aDst[h * 32 + k];
    }
    es[s] = a; ed[s] = d;
  }
  float* ob = M1 + (size_t)b * 9 * 256 + h * 32;
  for (int t = 0; t < 9; ++t) {
    float e[9], m = -1e30f;
#pragma unroll
    for (int s = 0; s < 9; ++s) {
      float x = es[s] + ed[t];
      e[s] = (x > 0.f) ? x : 0.2f * x;
      m = fmaxf(m, e[s]);
    }
    float w[9], den = 0.f;
#pragma unroll
    for (int s = 0; s < 9; ++s) { w[s] = expf(e[s] - m); den += w[s]; }
    float inv = 1.0f / den;
    for (int k = 0; k < 32; ++k) {
      float o = 0.f;
#pragma unroll
      for (int s = 0; s < 9; ++s) o += w[s] * base[s * 256 + k];
      o = o * inv + b1[h * 32 + k];
      ob[t * 256 + k] = fmaxf(o, 0.f);
    }
  }
}

// GAT layer-2 attention: only target node 0 is needed (cp). One thread / sample.
__global__ void gat2_attn(const float* __restrict__ XP2, const float* __restrict__ aSrc,
                          const float* __restrict__ aDst, const float* __restrict__ b2,
                          float* __restrict__ CP)
{
  int b = blockIdx.x * blockDim.x + threadIdx.x;
  if (b >= BTOT) return;
  const float* base = XP2 + (size_t)b * 9 * 256;
  float es[9], ed0 = 0.f;
#pragma unroll
  for (int s = 0; s < 9; ++s) {
    float a = 0.f;
    for (int k = 0; k < 256; ++k) a += base[s * 256 + k] * aSrc[k];
    es[s] = a;
  }
  for (int k = 0; k < 256; ++k) ed0 += base[k] * aDst[k];
  float e[9], m = -1e30f;
#pragma unroll
  for (int s = 0; s < 9; ++s) {
    float x = es[s] + ed0;
    e[s] = (x > 0.f) ? x : 0.2f * x;
    m = fmaxf(m, e[s]);
  }
  float w[9], den = 0.f;
#pragma unroll
  for (int s = 0; s < 9; ++s) { w[s] = expf(e[s] - m); den += w[s]; }
  float inv = 1.0f / den;
  for (int k = 0; k < 256; ++k) {
    float o = 0.f;
#pragma unroll
    for (int s = 0; s < 9; ++s) o += w[s] * base[s * 256 + k];
    CP[(size_t)b * 256 + k] = o * inv + b2[k];
  }
}

// fused = [txt*g, proj*(1-g), cp*g, adj(const)];  pred = sigmoid(fused @ fW + fb)
__global__ void final_kernel(const float* __restrict__ txt, const float* __restrict__ proj,
                             const float* __restrict__ cp, const float* __restrict__ g,
                             const float* __restrict__ fW, const float* __restrict__ fb,
                             float* __restrict__ pred)
{
  int b = blockIdx.x * blockDim.x + threadIdx.x;
  if (b >= BTOT) return;
  float acc = fb[0];
  size_t o = (size_t)b * 256;
  for (int j = 0; j < 256; ++j) {
    float gv = g[o + j];
    acc += txt[o + j]  * gv          * fW[j];
    acc += proj[o + j] * (1.f - gv)  * fW[256 + j];
    acc += cp[o + j]   * gv          * fW[512 + j];
  }
#pragma unroll
  for (int i = 0; i < 81; ++i)
    if ((i % 10) != 0) acc += fW[768 + i];   // adj = full graph minus diagonal
  pred[b] = 1.0f / (1.0f + expf(-acc));
}

// ---------------------------------------------------------------------------
extern "C" void kernel_launch(void* const* d_in, const int* in_sizes, int n_in,
                              void* d_out, int out_size, void* d_ws, size_t ws_size,
                              hipStream_t stream)
{
  (void)in_sizes; (void)n_in; (void)out_size; (void)ws_size;
  const float* sd_x   = (const float*)d_in[0];
  const float* ts_x   = (const float*)d_in[1];
  const float* text   = (const float*)d_in[2];
  /* d_in[3]: edge_index — fixed fully-connected graph, folded into kernels */
  const float* sd_W   = (const float*)d_in[4];
  const float* sd_b   = (const float*)d_in[5];
  const float* Wih_f  = (const float*)d_in[6];
  const float* Whh_f  = (const float*)d_in[7];
  const float* bih_f  = (const float*)d_in[8];
  const float* bhh_f  = (const float*)d_in[9];
  const float* Wih_b  = (const float*)d_in[10];
  const float* Whh_b  = (const float*)d_in[11];
  const float* bih_b  = (const float*)d_in[12];
  const float* bhh_b  = (const float*)d_in[13];
  const float* proj_W = (const float*)d_in[14];
  const float* proj_b = (const float*)d_in[15];
  const float* W1     = (const float*)d_in[16];
  const float* aS1    = (const float*)d_in[17];
  const float* aD1    = (const float*)d_in[18];
  const float* b1     = (const float*)d_in[19];
  const float* W2     = (const float*)d_in[20];
  const float* aS2    = (const float*)d_in[21];
  const float* aD2    = (const float*)d_in[22];
  const float* b2     = (const float*)d_in[23];
  const float* g_W    = (const float*)d_in[24];
  const float* g_b    = (const float*)d_in[25];
  const float* sdts_W = (const float*)d_in[26];
  const float* sdts_b = (const float*)d_in[27];
  const float* fW     = (const float*)d_in[28];
  const float* fb     = (const float*)d_in[29];

  float* out  = (float*)d_out;
  float* pred = out;                               // [B]
  float* proj = out + BTOT;                        // [B,256] (output #2)
  float* txt  = proj + (size_t)BTOT * 256;         // [B,256] (output #3)

  float* ws  = (float*)d_ws;
  float* sdH = ws;                                 // [B,256]
  float* tsH = sdH + (size_t)BTOT * 256;           // [B,256]
  float* sdN = tsH + (size_t)BTOT * 256;           // [B,256]
  float* XP  = sdN + (size_t)BTOT * 256;           // [B*9,256] (xp1, reused for xp2)
  float* M1  = XP  + (size_t)BTOT * 9 * 256;       // [B*9,256]
  float* G   = M1  + (size_t)BTOT * 9 * 256;       // [B,256]
  float* CP  = G   + (size_t)BTOT * 256;           // [B,256]
  bf16t* XPK = (bf16t*)(CP + (size_t)BTOT * 256);  // [256][48][2048] bf16 packed x

  dim3 blk(256);

  // 0) pack ts_x into WMMA-A blocks (one-time layout transform, bw-bound)
  pack_x_kernel<<<(256 * TSTEPS * 2048) / 256, blk, 0, stream>>>(ts_x, XPK);
  // 1) sd_h = gelu(sd_x @ sd_W + sd_b)
  gemm256_wmma<<<BTOT / 32, blk, 0, stream>>>(sd_x, 128, sd_W, 0, sd_b, sdH, 128, 1, 0, nullptr, nullptr, nullptr);
  // 2) sd_node = sd_h @ sdproj_W + b
  gemm256_wmma<<<BTOT / 32, blk, 0, stream>>>(sdH, 256, proj_W, 0, proj_b, sdN, 256, 0, 0, nullptr, nullptr, nullptr);
  // 3) GRU: forward = full 48 steps; backward contributes only cell(x[47], 0)
  gru_wmma<<<BTOT / 32, blk, 0, stream>>>(XPK, Wih_f, Whh_f, bih_f, bhh_f, tsH, 0, 0, 1, TSTEPS);
  gru_wmma<<<BTOT / 32, blk, 0, stream>>>(XPK, Wih_b, Whh_b, bih_b, bhh_b, tsH, 128, TSTEPS - 1, -1, 1);
  // 4) txt_rep (output #3, also feeds gate)
  txt_rep_kernel<<<(BTOT * 256) / 256, blk, 0, stream>>>(tsH, text, txt);
  // 5) g = sigmoid(txt_rep @ g_W + g_b)
  gemm256_wmma<<<BTOT / 32, blk, 0, stream>>>(txt, 256, g_W, 0, g_b, G, 256, 2, 0, nullptr, nullptr, nullptr);
  // 6) proj_sd_ts = concat(sd_h, ts_h) @ sdts_W + b  (output #2)
  gemm256_wmma<<<BTOT / 32, blk, 0, stream>>>(nullptr, 0, sdts_W, 0, sdts_b, proj, 512, 0, 2, sdH, tsH, nullptr);
  // 7) xp1 = xg @ W1 (xg gathered: node0=sd_node, node1=ts_h, 2..8=text_emb)
  gemm256_wmma<<<(BTOT * 9) / 32, blk, 0, stream>>>(nullptr, 0, W1, 0, nullptr, XP, 256, 0, 1, sdN, tsH, text);
  // 8) GAT layer-1 attention + bias + relu
  gat1_attn<<<(BTOT * 8) / 256, blk, 0, stream>>>(XP, aS1, aD1, b1, M1);
  // 9) xp2 = m1 @ W2 (reuse XP buffer)
  gemm256_wmma<<<(BTOT * 9) / 32, blk, 0, stream>>>(M1, 256, W2, 0, nullptr, XP, 256, 0, 0, nullptr, nullptr, nullptr);
  // 10) GAT layer-2 attention: cp = node-0 aggregation + b2
  gat2_attn<<<BTOT / 256, blk, 0, stream>>>(XP, aS2, aD2, b2, CP);
  // 11) gated fusion + final linear + sigmoid (output #1)
  final_kernel<<<BTOT / 256, blk, 0, stream>>>(txt, proj, CP, G, fW, fb, pred);
}
